// Attention_15264313770347
// MI455X (gfx1250) — matrix operand order repaired and checked by reading, not compiled, not run on hardware
//
#include <hip/hip_runtime.h>
#include <hip/hip_bf16.h>
#include <math.h>

// ---------------------------------------------------------------------------
// Bahdanau attention for MI455X (gfx1250, wave32, WMMA).
// Dominant GEMM (M=32768,N=1024,K=1024, 68.7 GFLOP) via
// v_wmma_f32_16x16x32_bf16, 2x2 register blocking (2 loads per WMMA).
// ---------------------------------------------------------------------------

#define HIDDEN 1024
#define BB 64
#define SS 512

typedef __attribute__((ext_vector_type(16))) __bf16 v16bf;
typedef __attribute__((ext_vector_type(8)))  __bf16 v8bf;
typedef __attribute__((ext_vector_type(8)))  float  v8f;

__device__ __forceinline__ unsigned short f32_to_bf16(float f) {
    union { float f; unsigned u; } c; c.f = f;
    unsigned u = c.u;
    u += 0x7FFFu + ((u >> 16) & 1u);   // round-to-nearest-even
    return (unsigned short)(u >> 16);
}

#define WMMA_BF16(a, b, c) \
    __builtin_amdgcn_wmma_f32_16x16x32_bf16(false, (a), false, (b), (short)0, (c), false, false)

// --------------------------- K0: fp32 -> bf16 ------------------------------
__global__ void __launch_bounds__(256)
convert_kernel(const float* __restrict__ W_h, const float* __restrict__ dech,
               unsigned short* __restrict__ Wb, unsigned short* __restrict__ dhb) {
    const int stride = gridDim.x * blockDim.x;
    for (int i = blockIdx.x * blockDim.x + threadIdx.x; i < HIDDEN * 2 * HIDDEN; i += stride)
        Wb[i] = f32_to_bf16(W_h[i]);
    for (int i = blockIdx.x * blockDim.x + threadIdx.x; i < BB * HIDDEN; i += stride)
        dhb[i] = f32_to_bf16(dech[i]);
}

// ------------- K1: dec_proj[b,h] = W_h[:, :H] @ dec_hidden -----------------
// GEMM M=64, N=1024, K=1024. grid = 4 M-tiles x 64 N-tiles, 1 wave each.
__global__ void __launch_bounds__(32)
decproj_kernel(const unsigned short* __restrict__ dhb,
               const unsigned short* __restrict__ Wb,
               float* __restrict__ dec_proj) {
    const int lane = threadIdx.x & 31;
    const int m    = lane & 15;
    const int half = lane >> 4;
    const int bm   = blockIdx.x >> 6;          // 0..3  (16 b-rows each)
    const int h0   = (blockIdx.x & 63) * 16;   // N tile
    const int n    = m;

    const unsigned short* arow = dhb + (size_t)(bm * 16 + m) * HIDDEN;
    const unsigned short* brow = Wb  + (size_t)(h0 + n) * (2 * HIDDEN); // K-half 0

    v8f acc = {};
    for (int k0 = 0; k0 < HIDDEN; k0 += 32) {
        v16bf a, bm16;
        v8bf* ah = (v8bf*)&a;
        // 16-bit A layout (ISA 7.12.2): lane half selects K octet groups
        ah[0] = *(const v8bf*)(arow + k0 + half * 8);
        ah[1] = *(const v8bf*)(arow + k0 + 16 + half * 8);
        // B 32x16: lane = column n, same K octet groups as A
        ((v8bf*)&bm16)[0] = *(const v8bf*)(brow + k0 + half * 8);
        ((v8bf*)&bm16)[1] = *(const v8bf*)(brow + k0 + 16 + half * 8);
        acc = WMMA_BF16(a, bm16, acc);
    }
    // C layout: lane holds column n, rows M = half*8 + j
    #pragma unroll
    for (int j = 0; j < 8; ++j)
        dec_proj[(size_t)(bm * 16 + half * 8 + j) * HIDDEN + h0 + n] = acc[j];
}

// -------- K2: scores[b,s] = v . tanh(enc @ W_enc^T + dec_proj[b]) ----------
// 1024 blocks (64 b x 16 s-tiles of 32 rows), 256 threads = 8 waves.
// A tile (32 s-rows x 1024 K) staged fp32->bf16 in LDS (64 KB of the 320 KB
// WGP pool), reused by all 64 N-tiles. Each wave register-blocks 2 M-subtiles
// x 2 N-tiles: 4 WMMAs per 8 b128 loads (2 loads/WMMA). B streams from the
// L2-resident 2 MB bf16 W_h copy.
__global__ void __launch_bounds__(256)
scores_kernel(const float* __restrict__ enc,
              const unsigned short* __restrict__ Wb,
              const float* __restrict__ dec_proj,
              const float* __restrict__ v,
              float* __restrict__ scores) {
    __shared__ __align__(16) unsigned short Atile[32 * HIDDEN]; // 64 KB
    __shared__ float spart[8][32];

    const int tid = threadIdx.x;
    const int b   = blockIdx.x >> 4;
    const int s0  = (blockIdx.x & 15) * 32;

    // ---- stage A tile: 32768 fp32 -> bf16, fully coalesced ----
    const float* src = enc + ((size_t)b * SS + s0) * HIDDEN;
    for (int i = tid; i < 32 * HIDDEN / 4; i += 256) {
        float4 f = ((const float4*)src)[i];
        uint2 pk;
        pk.x = (unsigned)f32_to_bf16(f.x) | ((unsigned)f32_to_bf16(f.y) << 16);
        pk.y = (unsigned)f32_to_bf16(f.z) | ((unsigned)f32_to_bf16(f.w) << 16);
        *(uint2*)&Atile[i * 4] = pk;
    }
    __syncthreads();

    const int wave = tid >> 5;
    const int lane = tid & 31;
    const int m    = lane & 15;
    const int half = lane >> 4;
    const int n    = m;

    // ssum[mt*8+j]: partial score for tile row mt*16 + half*8 + j
    float ssum[16];
    #pragma unroll
    for (int j = 0; j < 16; ++j) ssum[j] = 0.f;

    // 64 N-tiles = 32 adjacent pairs; each wave sweeps 4 pairs
    for (int w = 0; w < 4; ++w) {
        const int p  = wave + w * 8;         // pair index 0..31
        const int h0 = p * 32;               // first column of N-tile pair
        const unsigned short* brow0 =
            Wb + (size_t)(h0 + n) * (2 * HIDDEN) + HIDDEN;   // K-half 1 (encoder)
        const unsigned short* brow1 = brow0 + (size_t)16 * (2 * HIDDEN);

        v8f acc[2][2] = {{{}, {}}, {{}, {}}};  // [m-subtile][n-tile]
        for (int k0 = 0; k0 < HIDDEN; k0 += 32) {
            v16bf a0, a1, b0, b1;
            v8bf* a0h = (v8bf*)&a0; v8bf* a1h = (v8bf*)&a1;
            a0h[0] = *(const v8bf*)&Atile[(m     ) * HIDDEN + k0 + half * 8];
            a0h[1] = *(const v8bf*)&Atile[(m     ) * HIDDEN + k0 + 16 + half * 8];
            a1h[0] = *(const v8bf*)&Atile[(m + 16) * HIDDEN + k0 + half * 8];
            a1h[1] = *(const v8bf*)&Atile[(m + 16) * HIDDEN + k0 + 16 + half * 8];
            __builtin_prefetch(brow0 + k0 + 256, 0, 1);      // global_prefetch_b8
            __builtin_prefetch(brow1 + k0 + 256, 0, 1);
            ((v8bf*)&b0)[0] = *(const v8bf*)(brow0 + k0 + half * 8);
            ((v8bf*)&b0)[1] = *(const v8bf*)(brow0 + k0 + 16 + half * 8);
            ((v8bf*)&b1)[0] = *(const v8bf*)(brow1 + k0 + half * 8);
            ((v8bf*)&b1)[1] = *(const v8bf*)(brow1 + k0 + 16 + half * 8);
            acc[0][0] = WMMA_BF16(a0, b0, acc[0][0]);
            acc[1][0] = WMMA_BF16(a1, b0, acc[1][0]);
            acc[0][1] = WMMA_BF16(a0, b1, acc[0][1]);
            acc[1][1] = WMMA_BF16(a1, b1, acc[1][1]);
        }
        // epilogue: tanh + dot with v, reduce across the 16 columns per tile
        #pragma unroll
        for (int nt = 0; nt < 2; ++nt) {
            const float dpv = dec_proj[(size_t)b * HIDDEN + h0 + nt * 16 + n];
            const float vn  = v[h0 + nt * 16 + n];
            #pragma unroll
            for (int mt = 0; mt < 2; ++mt) {
                #pragma unroll
                for (int j = 0; j < 8; ++j) {
                    float t = vn * tanhf(acc[mt][nt][j] + dpv);
                    #pragma unroll
                    for (int off = 1; off < 16; off <<= 1)  // within 16-lane half
                        t += __shfl_xor(t, off, 32);
                    ssum[mt * 8 + j] += t;                  // valid on lanes m==0
                }
            }
        }
    }
    if (m == 0) {
        #pragma unroll
        for (int mt = 0; mt < 2; ++mt)
            #pragma unroll
            for (int j = 0; j < 8; ++j)
                spart[wave][mt * 16 + half * 8 + j] = ssum[mt * 8 + j];
    }
    __syncthreads();
    if (tid < 32) {                                  // deterministic cross-wave sum
        float s = 0.f;
        #pragma unroll
        for (int wv = 0; wv < 8; ++wv) s += spart[wv][tid];
        scores[(size_t)b * SS + s0 + tid] = s;
    }
}

// ------------- K3: softmax over batch dim (axis=0) + mask ------------------
__global__ void __launch_bounds__(64)
softmax_b_kernel(const float* __restrict__ scores, const float* __restrict__ mask,
                 float* __restrict__ p) {
    __shared__ float red[64];
    const int s = blockIdx.x, b = threadIdx.x;
    const float x = scores[(size_t)b * SS + s];
    red[b] = x; __syncthreads();
    for (int off = 32; off > 0; off >>= 1) {
        if (b < off) red[b] = fmaxf(red[b], red[b + off]);
        __syncthreads();
    }
    const float mx = red[0]; __syncthreads();
    const float e = expf(x - mx);
    red[b] = e; __syncthreads();
    for (int off = 32; off > 0; off >>= 1) {
        if (b < off) red[b] += red[b + off];
        __syncthreads();
    }
    p[(size_t)b * SS + s] = (e / red[0]) * mask[(size_t)s * BB + b]; // mask is [S,B]
}

// --------- K4: renormalize over S, emit a, context = a @ enc ---------------
__global__ void __launch_bounds__(256)
context_kernel(const float* __restrict__ p, const float* __restrict__ enc,
               float* __restrict__ a_out, float* __restrict__ ctx_out) {
    __shared__ float a_lds[SS];
    __shared__ float red[256];
    const int b = blockIdx.x, tid = threadIdx.x;

    const float p0 = p[(size_t)b * SS + tid];
    const float p1 = p[(size_t)b * SS + tid + 256];
    red[tid] = p0 + p1; __syncthreads();
    for (int off = 128; off > 0; off >>= 1) {
        if (tid < off) red[tid] += red[tid + off];
        __syncthreads();
    }
    const float inv = 1.f / (red[0] + 1e-10f);
    const float a0 = p0 * inv, a1 = p1 * inv;
    a_out[(size_t)b * SS + tid]       = a0;
    a_out[(size_t)b * SS + tid + 256] = a1;
    a_lds[tid] = a0; a_lds[tid + 256] = a1;
    __syncthreads();

    const float* erow = enc + (size_t)b * SS * HIDDEN;
    const int h0 = tid * 4;
    float4 acc = {0.f, 0.f, 0.f, 0.f};
    for (int s = 0; s < SS; ++s) {
        const float as = a_lds[s];
        const float4 ev = *(const float4*)(erow + (size_t)s * HIDDEN + h0);
        acc.x += as * ev.x; acc.y += as * ev.y;
        acc.z += as * ev.z; acc.w += as * ev.w;
    }
    *(float4*)(ctx_out + (size_t)b * HIDDEN + h0) = acc;
}

// ---------------------------------------------------------------------------
extern "C" void kernel_launch(void* const* d_in, const int* in_sizes, int n_in,
                              void* d_out, int out_size, void* d_ws, size_t ws_size,
                              hipStream_t stream) {
    const float* enc  = (const float*)d_in[0];   // [64,512,1024]
    const float* dech = (const float*)d_in[1];   // [64,1024]
    const float* mask = (const float*)d_in[2];   // [512,64]
    const float* W_h  = (const float*)d_in[3];   // [1024,2048]
    const float* v    = (const float*)d_in[4];   // [1024]
    float* out = (float*)d_out;                  // a (64*512) ++ context (64*1024)

    char* ws = (char*)d_ws;                      // ~4.6 MB total
    unsigned short* Wb       = (unsigned short*)(ws);                          // 4 MB
    unsigned short* dhb      = (unsigned short*)(ws + (4u << 20));             // 128 KB
    float*          dec_proj = (float*)(ws + (4u << 20) + (128u << 10));       // 256 KB
    float*          scores   = (float*)(ws + (4u << 20) + (384u << 10));       // 128 KB
    float*          p        = (float*)(ws + (4u << 20) + (512u << 10));       // 128 KB

    convert_kernel  <<<1024, 256, 0, stream>>>(W_h, dech, Wb, dhb);
    decproj_kernel  <<<256,   32, 0, stream>>>(dhb, Wb, dec_proj);
    scores_kernel   <<<1024, 256, 0, stream>>>(enc, Wb, dec_proj, v, scores);
    softmax_b_kernel<<<SS,    64, 0, stream>>>(scores, mask, p);
    context_kernel  <<<BB,   256, 0, stream>>>(p, enc, out, out + BB * SS);
}
